// SpatioConvLayer_35296041239233
// MI455X (gfx1250) — matrix operand order repaired and checked by reading, not compiled, not run on hardware
//
#include <hip/hip_runtime.h>

typedef __attribute__((ext_vector_type(16))) _Float16 v16h;
typedef __attribute__((ext_vector_type(8)))  float    v8f;

#define Bsz  8
#define Nn   8192
#define Cc   32
#define KSn  3
#define KTOT (KSn * Nn)      // 24576  (GEMM K dimension, (k,m) fused)
#define COLS (Bsz * Cc)      // 256    (GEMM N dimension, (b,o) fused)

// ---------------------------------------------------------------------------
// Pre-pass: Yt[col=(b*32+o)][kk=k*8192+m] = sum_i x[b,m,i] * theta[i,o,k]
// Stored f16, K-contiguous per column -> WMMA B fragments are one contiguous
// 32B load per lane. Total Yt = 256*24576*2B = 12.6 MB (L2-resident).
// ---------------------------------------------------------------------------
__global__ __launch_bounds__(256) void prep_kernel(const float* __restrict__ x,
                                                   const float* __restrict__ theta,
                                                   _Float16* __restrict__ Yt) {
    __shared__ float th_s[Cc * Cc * KSn];   // theta[i][o][k] flat, 12 KB
    __shared__ float xs[16 * COLS];         // xs[mm][b*32+i], 16 KB

    const int tid = threadIdx.x;
    for (int idx = tid; idx < Cc * Cc * KSn; idx += 256) th_s[idx] = theta[idx];

    const int m0 = blockIdx.x * 16;
    const int b  = tid >> 5;
    const int io = tid & 31;
    #pragma unroll
    for (int mm = 0; mm < 16; ++mm)
        xs[mm * COLS + tid] = x[((size_t)b * Nn + m0 + mm) * Cc + io];
    __syncthreads();

    const int o = io;
    for (int k = 0; k < KSn; ++k) {
        v16h h;
        #pragma unroll
        for (int mm = 0; mm < 16; ++mm) {
            float acc = 0.0f;
            #pragma unroll
            for (int i = 0; i < Cc; ++i)
                acc += xs[mm * COLS + (b << 5) + i] * th_s[(i * Cc + o) * KSn + k];
            h[mm] = (_Float16)acc;
        }
        *(v16h*)(Yt + (size_t)tid * KTOT + (size_t)k * Nn + m0) = h;
    }
}

// ---------------------------------------------------------------------------
// Main GEMM: D[n, col] = sum_kk A[n,kk] * B[kk,col]
//   A[n, k*8192+m] = Lk[k][n][m]   (f32 in HBM, converted to f16 in-register)
//   B = Yt (f16, L2-resident)
// Fused epilogue: + bias[o] + x[b,n,o], ReLU, store out[b,n,o].
// Block: 256 threads = 8 waves = 4 row-groups x 2 col-groups.
// Wave tile: 16 rows x 128 cols (8 WMMA accumulators). Block tile: 64 x 256.
// ---------------------------------------------------------------------------
__global__ __launch_bounds__(256) void gemm_kernel(const float* __restrict__ Lk,
                                                   const _Float16* __restrict__ Yt,
                                                   const float* __restrict__ x,
                                                   const float* __restrict__ bias,
                                                   float* __restrict__ out) {
    const int lane     = threadIdx.x & 31;
    const int wave     = threadIdx.x >> 5;     // 0..7
    const int rowGroup = wave >> 1;            // 0..3
    const int colGroup = wave & 1;             // 0..1
    const int l15      = lane & 15;
    const int hi       = (lane >> 4) & 1;      // half-wave selector
    const int rowBase  = blockIdx.x * 64 + rowGroup * 16;
    const int aRow     = rowBase + l15;        // A: lane = matrix row M
    const int klo      = hi ? 16 : 0;          // K offset: lanes>=16 take K 16..31 (contiguous order)
    const int colBase  = colGroup * 128;

    v8f zero = {};
    v8f acc[8];
    #pragma unroll
    for (int t = 0; t < 8; ++t) acc[t] = zero;

    // B fragment base: lane l15 -> column colBase + t*16 + l15; lanes>=16 read K+16
    const _Float16* Bbase = Yt + (size_t)(colBase + l15) * KTOT + hi * 16;

    for (int k = 0; k < KSn; ++k) {
        const float*    Ar = Lk + (size_t)k * Nn * Nn + (size_t)aRow * Nn + klo;
        const _Float16* Bk = Bbase + (size_t)k * Nn;
        for (int m0 = 0; m0 < Nn; m0 += 32) {
            __builtin_prefetch(Ar + m0 + 512, 0, 0);   // stream Lk ahead (HBM)
            // A fragment: lane<16 holds K {0..15}, lane>=16 holds K {16..31}
            const float4 a0 = *(const float4*)(Ar + m0);
            const float4 a1 = *(const float4*)(Ar + m0 + 4);
            const float4 a2 = *(const float4*)(Ar + m0 + 8);
            const float4 a3 = *(const float4*)(Ar + m0 + 12);
            v16h a;
            a[0]  = (_Float16)a0.x;  a[1]  = (_Float16)a0.y;
            a[2]  = (_Float16)a0.z;  a[3]  = (_Float16)a0.w;
            a[4]  = (_Float16)a1.x;  a[5]  = (_Float16)a1.y;
            a[6]  = (_Float16)a1.z;  a[7]  = (_Float16)a1.w;
            a[8]  = (_Float16)a2.x;  a[9]  = (_Float16)a2.y;
            a[10] = (_Float16)a2.z;  a[11] = (_Float16)a2.w;
            a[12] = (_Float16)a3.x;  a[13] = (_Float16)a3.y;
            a[14] = (_Float16)a3.z;  a[15] = (_Float16)a3.w;

            const _Float16* Bm = Bk + m0;
            #pragma unroll
            for (int t = 0; t < 8; ++t) {
                v16h bf = *(const v16h*)(Bm + (size_t)t * 16 * KTOT);
                acc[t] = __builtin_amdgcn_wmma_f32_16x16x32_f16(
                    false, a, false, bf, (short)0, acc[t], false, false);
            }
        }
    }

    // Epilogue: D VGPR g, lane L -> row = rowBase + g + hi*8, col = tile*16 + l15
    const float biasA = bias[l15];
    const float biasB = bias[16 + l15];
    #pragma unroll
    for (int t = 0; t < 8; ++t) {
        const int col = colBase + t * 16 + l15;
        const int bb  = col >> 5;
        const int o   = col & 31;
        const float bs = (t & 1) ? biasB : biasA;
        #pragma unroll
        for (int g = 0; g < 8; ++g) {
            const int n = rowBase + g + hi * 8;
            const size_t idx = ((size_t)bb * Nn + n) * Cc + o;
            const float v = acc[t][g] + bs + x[idx];
            out[idx] = v > 0.0f ? v : 0.0f;
        }
    }
}

// ---------------------------------------------------------------------------
extern "C" void kernel_launch(void* const* d_in, const int* in_sizes, int n_in,
                              void* d_out, int out_size, void* d_ws, size_t ws_size,
                              hipStream_t stream) {
    const float* x     = (const float*)d_in[0];   // (8, 8192, 32)
    const float* Lk    = (const float*)d_in[1];   // (3, 8192, 8192)
    const float* theta = (const float*)d_in[2];   // (32, 32, 3)
    const float* bias  = (const float*)d_in[3];   // (1, 32, 1) -> 32 floats
    float*       out   = (float*)d_out;           // (8, 8192, 32)
    _Float16*    Yt    = (_Float16*)d_ws;         // 256 * 24576 f16 = 12.6 MB

    prep_kernel<<<Nn / 16, 256, 0, stream>>>(x, theta, Yt);
    gemm_kernel<<<Nn / 64, 256, 0, stream>>>(Lk, Yt, x, bias, out);
}
